// GCViTBlock_3135326126801
// MI455X (gfx1250) — compile-verified, matrix operand order checked by reading, not run on hardware
//
#include <hip/hip_runtime.h>
#include <cstdint>

// ---------------------------------------------------------------------------
// Types / helpers for CDNA5 WMMA (wave32, 16x16x32 bf16 -> f32)
// ---------------------------------------------------------------------------
typedef __attribute__((ext_vector_type(16))) __bf16 v16bf;
typedef __attribute__((ext_vector_type(8)))  float  v8f;

__device__ __forceinline__ unsigned short f2b(float f) {
    unsigned x = __builtin_bit_cast(unsigned, f);
    x += 0x7FFFu + ((x >> 16) & 1u);           // round-to-nearest-even
    return (unsigned short)(x >> 16);
}

struct Frag {
    union { v16bf v; uint4 q[2]; };
};
static_assert(sizeof(Frag) == 32, "frag size");

// Load a 16x32 bf16 fragment (A-layout; B-layout is the same pattern with the
// tile stored [N][K]) from LDS. ISA layout: lanes 0-15 rows M=0..15 hold K
// pairs {0,1},{2,3},{4,5},{6,7},{16,17},{18,19},{20,21},{22,23}; lanes 16-31
// hold the +8 K pairs.  With `stride` a multiple of 8 elements, both halves
// are single 16-byte ds_load_b128s.
__device__ __forceinline__ Frag ldfrag(const unsigned short* s, int row0,
                                       int stride, int koff, int lane) {
    Frag f;
    const int ml = lane & 15, half = lane >> 4;
    const unsigned short* p = s + (size_t)(row0 + ml) * stride + koff + half * 8;
    f.q[0] = *(const uint4*)(p);        // K elements 0..7   (of this half)
    f.q[1] = *(const uint4*)(p + 16);   // K elements 16..23 (of this half)
    return f;
}

__device__ __forceinline__ v8f wmma_bf16(const Frag& a, const Frag& b, v8f c) {
    return __builtin_amdgcn_wmma_f32_16x16x32_bf16(
        false, a.v, false, b.v, (short)0, c, false, false);
}

// ---- VALU-only 16-lane butterfly reductions via DPP16 (no LDS, no waits) ----
template <int CTRL>
__device__ __forceinline__ float dpp_mov(float x) {
    int i = __builtin_bit_cast(int, x);
    i = __builtin_amdgcn_update_dpp(i, i, CTRL, 0xF, 0xF, false);
    return __builtin_bit_cast(float, i);
}

__device__ __forceinline__ float red16_max(float v) {
    v = fmaxf(v, dpp_mov<0xB1>(v));    // quad_perm(1,0,3,2) : xor 1
    v = fmaxf(v, dpp_mov<0x4E>(v));    // quad_perm(2,3,0,1) : xor 2
    v = fmaxf(v, dpp_mov<0x141>(v));   // row_half_mirror    : swap quads in 8
    v = fmaxf(v, dpp_mov<0x140>(v));   // row_mirror         : swap 8s in 16
    return v;
}

__device__ __forceinline__ float red16_sum(float v) {
    v += dpp_mov<0xB1>(v);
    v += dpp_mov<0x4E>(v);
    v += dpp_mov<0x141>(v);
    v += dpp_mov<0x140>(v);
    return v;
}

// ---------------------------------------------------------------------------
// Small elementwise converters
// ---------------------------------------------------------------------------
__global__ void cvt_bf16_kernel(const float* __restrict__ s,
                                unsigned short* __restrict__ d, int n) {
    int i = blockIdx.x * 256 + threadIdx.x;
    if (i < n) d[i] = f2b(s[i]);
}

__global__ void cvt_scale_kernel(const float* __restrict__ s,
                                 unsigned short* __restrict__ d, int n, float k) {
    int i = blockIdx.x * 256 + threadIdx.x;
    if (i < n) d[i] = f2b(s[i] * k);
}

// Densify relative-position bias into [NH=16][64][64] f32 with key-padding
// mask baked in, so the attention hot loop is a plain vector load + add.
__global__ void bias_tab_kernel(const float* __restrict__ relTab,  // [169][16]
                                float* __restrict__ bt) {          // [16][64][64]
    const int i = blockIdx.x * 256 + threadIdx.x;   // 65536 total
    if (i >= 16 * 64 * 64) return;
    const int col = i & 63;
    const int row = (i >> 6) & 63;
    const int h = i >> 12;
    float v;
    if (col >= 49) {
        v = -1e30f;                                  // masked (padded key)
    } else if (row >= 49) {
        v = 0.0f;                                    // padded query row: benign
    } else {
        const int dr = row / 7 - col / 7 + 6;
        const int dc = row % 7 - col % 7 + 6;
        v = relTab[(dr * 13 + dc) * 16 + h];
    }
    bt[i] = v;
}

// ---------------------------------------------------------------------------
// LayerNorm over C=512, one block (256 threads) per row.
//   ln_win : output window-partitioned bf16 [B_=512][N=49][C]
//   ln_plain: output plain-order bf16 [25088][C]
// ---------------------------------------------------------------------------
__device__ __forceinline__ float block_sum_512(float s, float* sh, int t) {
    #pragma unroll
    for (int o = 16; o > 0; o >>= 1) s += __shfl_xor(s, o);
    if ((t & 31) == 0) sh[t >> 5] = s;
    __syncthreads();
    float tot = 0.f;
    #pragma unroll
    for (int i = 0; i < 8; ++i) tot += sh[i];
    __syncthreads();
    return tot;
}

__global__ __launch_bounds__(256) void ln_win_kernel(
    const float* __restrict__ x, const float* __restrict__ w,
    const float* __restrict__ b, unsigned short* __restrict__ out) {
    __shared__ float sh[8];
    const int row = blockIdx.x;            // b*784 + hh*28 + ww
    const int t = threadIdx.x;
    const float* xr = x + (size_t)row * 512;
    const float v0 = xr[t], v1 = xr[t + 256];

    const float mean = block_sum_512(v0 + v1, sh, t) * (1.0f / 512.0f);
    const float d0 = v0 - mean, d1 = v1 - mean;
    const float var = block_sum_512(d0 * d0 + d1 * d1, sh, t) * (1.0f / 512.0f);
    const float rstd = rsqrtf(var + 1e-5f);

    const float y0 = d0 * rstd * w[t] + b[t];
    const float y1 = d1 * rstd * w[t + 256] + b[t + 256];

    const int bb = row / 784, rem = row % 784;
    const int hh = rem / 28, ww = rem % 28;
    const int b_ = (bb * 4 + hh / 7) * 4 + ww / 7;
    const int n = (hh % 7) * 7 + (ww % 7);
    const size_t o = ((size_t)b_ * 49 + n) * 512;
    out[o + t] = f2b(y0);
    out[o + t + 256] = f2b(y1);
}

__global__ __launch_bounds__(256) void ln_plain_kernel(
    const float* __restrict__ x, const float* __restrict__ w,
    const float* __restrict__ b, unsigned short* __restrict__ out) {
    __shared__ float sh[8];
    const int row = blockIdx.x;
    const int t = threadIdx.x;
    const float* xr = x + (size_t)row * 512;
    const float v0 = xr[t], v1 = xr[t + 256];

    const float mean = block_sum_512(v0 + v1, sh, t) * (1.0f / 512.0f);
    const float d0 = v0 - mean, d1 = v1 - mean;
    const float var = block_sum_512(d0 * d0 + d1 * d1, sh, t) * (1.0f / 512.0f);
    const float rstd = rsqrtf(var + 1e-5f);

    const size_t o = (size_t)row * 512;
    out[o + t] = f2b(d0 * rstd * w[t] + b[t]);
    out[o + t + 256] = f2b(d1 * rstd * w[t + 256] + b[t + 256]);
}

// ---------------------------------------------------------------------------
// Generic bf16 WMMA GEMM:  out[m,n] = sum_k A[m,k] * B[n,k]  (B = weight [out,in])
// Block tile 128x128, K-step 32, 8 waves of 64x32, double-buffered LDS.
// Epilogues fused per stage.
// ---------------------------------------------------------------------------
enum { EPI_KV = 0, EPI_PROJ = 1, EPI_GELU = 2, EPI_OUT = 3 };

template <int EPI>
__global__ __launch_bounds__(256) void gemm_bf16_kernel(
    const unsigned short* __restrict__ A,   // [M x K] bf16, M = 25088
    const unsigned short* __restrict__ B,   // [Nout x K] bf16
    const int K, const int Nout,
    const float* __restrict__ bias,         // [Nout]
    const float* __restrict__ addsrc,       // residual source (f32) or null
    float* __restrict__ outF,
    unsigned short* __restrict__ outB,
    unsigned short* __restrict__ outKm,
    unsigned short* __restrict__ outVm) {
    __shared__ unsigned short sA[2][128 * 40];  // stride 40: 16B-aligned rows,
    __shared__ unsigned short sB[2][128 * 40];  // conflict-free fragment reads

    const int tid = threadIdx.x;
    const int lane = tid & 31;
    const int wave = tid >> 5;
    const int waveM = wave >> 2;   // 0..1  -> 64-row band
    const int waveN = wave & 3;    // 0..3  -> 32-col band
    const int ml = lane & 15, half = lane >> 4;

    const int mBase = blockIdx.x * 128;
    const int nBase = blockIdx.y * 128;

    // cooperative global->LDS: each thread owns 16 contiguous bf16 of one row
    const int ldRow = tid >> 1;           // 0..127
    const int ldCol = (tid & 1) * 16;     // 0 / 16
    const unsigned short* gA = A + (size_t)(mBase + ldRow) * K + ldCol;
    const unsigned short* gB = B + (size_t)(nBase + ldRow) * K + ldCol;
    const int loff = ldRow * 40 + ldCol;

    uint4 ra0 = *(const uint4*)(gA);
    uint4 ra1 = *(const uint4*)(gA + 8);
    uint4 rb0 = *(const uint4*)(gB);
    uint4 rb1 = *(const uint4*)(gB + 8);
    *(uint4*)&sA[0][loff] = ra0;
    *(uint4*)&sA[0][loff + 8] = ra1;
    *(uint4*)&sB[0][loff] = rb0;
    *(uint4*)&sB[0][loff + 8] = rb1;
    __syncthreads();

    v8f acc[4][2];
    #pragma unroll
    for (int tm = 0; tm < 4; ++tm)
        #pragma unroll
        for (int tn = 0; tn < 2; ++tn)
            #pragma unroll
            for (int r = 0; r < 8; ++r) acc[tm][tn][r] = 0.0f;

    const int nk = K >> 5;
    for (int kb = 0; kb < nk; ++kb) {
        const int cur = kb & 1;
        if (kb + 1 < nk) {  // register prefetch of next K-block
            const unsigned short* pa = gA + (size_t)(kb + 1) * 32;
            const unsigned short* pb = gB + (size_t)(kb + 1) * 32;
            ra0 = *(const uint4*)(pa);
            ra1 = *(const uint4*)(pa + 8);
            rb0 = *(const uint4*)(pb);
            rb1 = *(const uint4*)(pb + 8);
        }
        Frag fa[4], fb[2];
        #pragma unroll
        for (int tm = 0; tm < 4; ++tm)
            fa[tm] = ldfrag(sA[cur], waveM * 64 + tm * 16, 40, 0, lane);
        #pragma unroll
        for (int tn = 0; tn < 2; ++tn)
            fb[tn] = ldfrag(sB[cur], waveN * 32 + tn * 16, 40, 0, lane);
        #pragma unroll
        for (int tm = 0; tm < 4; ++tm)
            #pragma unroll
            for (int tn = 0; tn < 2; ++tn)
                acc[tm][tn] = wmma_bf16(fa[tm], fb[tn], acc[tm][tn]);
        if (kb + 1 < nk) {
            __syncthreads();
            const int nxt = cur ^ 1;
            *(uint4*)&sA[nxt][loff] = ra0;
            *(uint4*)&sA[nxt][loff + 8] = ra1;
            *(uint4*)&sB[nxt][loff] = rb0;
            *(uint4*)&sB[nxt][loff + 8] = rb1;
            __syncthreads();
        }
    }

    // ------------------------------ epilogue ------------------------------
    #pragma unroll
    for (int tm = 0; tm < 4; ++tm) {
        #pragma unroll
        for (int tn = 0; tn < 2; ++tn) {
            const int c = nBase + waveN * 32 + tn * 16 + ml;
            const float bc = bias[c];
            #pragma unroll
            for (int r = 0; r < 8; ++r) {
                const int m = mBase + waveM * 64 + tm * 16 + r + half * 8;
                const float v = acc[tm][tn][r] + bc;
                if constexpr (EPI == EPI_KV) {
                    // scatter to K/V [B_][NH][N][hd] bf16
                    const int b_ = m / 49, n = m % 49;
                    const int pair = c >> 9, head = (c >> 5) & 15, d = c & 31;
                    const size_t di = ((size_t)(b_ * 16 + head) * 49 + n) * 32 + d;
                    if (pair) outVm[di] = f2b(v);
                    else      outKm[di] = f2b(v);
                } else if constexpr (EPI == EPI_PROJ) {
                    // window-reverse + residual -> X1 f32 [B,H,W,C]
                    const int b_ = m / 49, n = m % 49;
                    const int bb = b_ >> 4, win = b_ & 15;
                    const int hh = (win >> 2) * 7 + n / 7;
                    const int ww = (win & 3) * 7 + n % 7;
                    const size_t gi = (((size_t)bb * 28 + hh) * 28 + ww) * 512 + c;
                    outF[gi] = addsrc[gi] + v;
                } else if constexpr (EPI == EPI_GELU) {
                    const float g = 0.5f * v * (1.0f + erff(v * 0.70710678118f));
                    outB[(size_t)m * Nout + c] = f2b(g);
                } else {  // EPI_OUT: final residual
                    const size_t gi = (size_t)m * Nout + c;
                    outF[gi] = addsrc[gi] + v;
                }
            }
        }
    }
}

// ---------------------------------------------------------------------------
// Windowed attention, one block = one (window, head). 128 threads = 4 waves,
// each wave owns a 16-row band of the 64(pad of 49)x64 attention matrix.
//   attn = (q*scale) k^T + biasTab ; softmax ; out = attn v
// biasTab already contains the -1e30 key-padding mask -> branch-free epilogue.
// Softmax row reductions are DPP16 butterflies (VALU only).
// ---------------------------------------------------------------------------
__global__ __launch_bounds__(128) void attn_kernel(
    const unsigned short* __restrict__ Q,    // [32][16][49][32] bf16 (prescaled)
    const unsigned short* __restrict__ Km,   // [512][16][49][32]
    const unsigned short* __restrict__ Vm,   // [512][16][49][32]
    const float* __restrict__ biasTab,       // [16][64][64] f32, mask baked in
    unsigned short* __restrict__ AO) {       // [512*49][512] bf16
    const int blk = blockIdx.x;
    const int b_ = blk >> 4;                 // window 0..511
    const int h = blk & 15;                  // head
    const int tid = threadIdx.x;
    const int lane = tid & 31;
    const int wave = tid >> 5;               // band 0..3
    const int ml = lane & 15, half = lane >> 4;

    __shared__ unsigned short sq[64 * 40];   // q rows (zero-padded to 64)
    __shared__ unsigned short sk[64 * 40];   // k rows
    __shared__ unsigned short sv[32 * 72];   // v transposed: [d][m]
    __shared__ unsigned short sp[64 * 72];   // softmax probs (A for P@V)

    const unsigned short* qbase = Q + ((size_t)((b_ >> 4) * 16 + h) * 49) * 32;
    const unsigned short* kbase = Km + ((size_t)(b_ * 16 + h) * 49) * 32;
    const unsigned short* vbase = Vm + ((size_t)(b_ * 16 + h) * 49) * 32;

    for (int i = tid; i < 64 * 32; i += 128) {
        const int m = i >> 5, d = i & 31;
        unsigned short qv = 0, kv = 0, vv = 0;
        if (m < 49) {
            qv = qbase[m * 32 + d];
            kv = kbase[m * 32 + d];
            vv = vbase[m * 32 + d];
        }
        sq[m * 40 + d] = qv;
        sk[m * 40 + d] = kv;
        sv[d * 72 + m] = vv;   // transpose so P@V B-frags read contiguous pairs
    }
    __syncthreads();

    // ---- QK^T : 4 WMMAs per band ----
    const Frag aq = ldfrag(sq, wave * 16, 40, 0, lane);
    v8f att[4];
    #pragma unroll
    for (int tn = 0; tn < 4; ++tn) {
        #pragma unroll
        for (int r = 0; r < 8; ++r) att[tn][r] = 0.0f;
        const Frag bk = ldfrag(sk, tn * 16, 40, 0, lane);
        att[tn] = wmma_bf16(aq, bk, att[tn]);
    }

    // ---- add dense (pre-masked) bias: branch-free coalesced loads ----
    // element (tn, r) -> row = wave*16 + r + half*8, col = tn*16 + ml
    const float* bt = biasTab + ((size_t)h << 12) + (half << 9) + ml;
    #pragma unroll
    for (int tn = 0; tn < 4; ++tn) {
        const float* btc = bt + (size_t)wave * 16 * 64 + tn * 16;
        #pragma unroll
        for (int r = 0; r < 8; ++r) {
            att[tn][r] += btc[r * 64];
        }
    }

    // ---- softmax per query row (cols live in a 16-lane group x 4 tiles) ----
    #pragma unroll
    for (int r = 0; r < 8; ++r) {
        float mx = fmaxf(fmaxf(att[0][r], att[1][r]), fmaxf(att[2][r], att[3][r]));
        mx = red16_max(mx);
        float s = 0.f;
        #pragma unroll
        for (int tn = 0; tn < 4; ++tn) {
            const float e = __expf(att[tn][r] - mx);
            att[tn][r] = e;
            s += e;
        }
        s = red16_sum(s);
        const float rs = 1.0f / s;
        #pragma unroll
        for (int tn = 0; tn < 4; ++tn) {
            const int row = wave * 16 + r + half * 8;
            sp[row * 72 + tn * 16 + ml] = f2b(att[tn][r] * rs);
        }
    }
    // band-local LDS reuse: producer wave == consumer wave, no barrier needed.

    // ---- P @ V : K = 64 (2 steps), N = 32 (2 tiles) ----
    #pragma unroll
    for (int tn2 = 0; tn2 < 2; ++tn2) {
        v8f o;
        #pragma unroll
        for (int r = 0; r < 8; ++r) o[r] = 0.0f;
        #pragma unroll
        for (int ks = 0; ks < 2; ++ks) {
            const Frag ap = ldfrag(sp, wave * 16, 72, ks * 32, lane);
            const Frag bv = ldfrag(sv, tn2 * 16, 72, ks * 32, lane);
            o = wmma_bf16(ap, bv, o);
        }
        #pragma unroll
        for (int r = 0; r < 8; ++r) {
            const int row = wave * 16 + r + half * 8;
            if (row < 49) {
                AO[((size_t)b_ * 49 + row) * 512 + h * 32 + tn2 * 16 + ml] = f2b(o[r]);
            }
        }
    }
}

// ---------------------------------------------------------------------------
// Launch: orchestrate the whole block. Workspace bump-allocated (~265 MB).
// ---------------------------------------------------------------------------
extern "C" void kernel_launch(void* const* d_in, const int* in_sizes, int n_in,
                              void* d_out, int out_size, void* d_ws, size_t ws_size,
                              hipStream_t stream) {
    (void)in_sizes; (void)n_in; (void)out_size; (void)ws_size;
    const float* x      = (const float*)d_in[0];
    const float* qg     = (const float*)d_in[1];
    const float* n1w    = (const float*)d_in[2];
    const float* n1b    = (const float*)d_in[3];
    const float* qkv_w  = (const float*)d_in[4];
    const float* qkv_b  = (const float*)d_in[5];
    const float* relTab = (const float*)d_in[6];
    const float* proj_w = (const float*)d_in[7];
    const float* proj_b = (const float*)d_in[8];
    const float* n2w    = (const float*)d_in[9];
    const float* n2b    = (const float*)d_in[10];
    const float* fc1_w  = (const float*)d_in[11];
    const float* fc1_b  = (const float*)d_in[12];
    const float* fc2_w  = (const float*)d_in[13];
    const float* fc2_b  = (const float*)d_in[14];
    float* out = (float*)d_out;

    char* ws = (char*)d_ws;
    size_t off = 0;
    auto alloc = [&](size_t bytes) -> void* {
        off = (off + 255) & ~(size_t)255;
        void* p = ws + off;
        off += bytes;
        return p;
    };

    unsigned short* Wqkv  = (unsigned short*)alloc((size_t)1024 * 512 * 2);
    unsigned short* Wproj = (unsigned short*)alloc((size_t)512 * 512 * 2);
    unsigned short* Wf1   = (unsigned short*)alloc((size_t)2048 * 512 * 2);
    unsigned short* Wf2   = (unsigned short*)alloc((size_t)512 * 2048 * 2);
    unsigned short* Q16   = (unsigned short*)alloc((size_t)32 * 16 * 49 * 32 * 2);
    float*          BT    = (float*)alloc((size_t)16 * 64 * 64 * 4);
    unsigned short* XN    = (unsigned short*)alloc((size_t)25088 * 512 * 2);  // ln1 out; reused as H2
    unsigned short* K16   = (unsigned short*)alloc((size_t)512 * 16 * 49 * 32 * 2);
    unsigned short* V16   = (unsigned short*)alloc((size_t)512 * 16 * 49 * 32 * 2);
    unsigned short* AO    = (unsigned short*)alloc((size_t)25088 * 512 * 2);
    float*          X1    = (float*)alloc((size_t)25088 * 512 * 4);
    unsigned short* M1    = (unsigned short*)alloc((size_t)25088 * 2048 * 2);

    // 1) weights / query to bf16 (q prescaled by hd^-0.5); dense bias table
    cvt_bf16_kernel<<<(1024 * 512 + 255) / 256, 256, 0, stream>>>(qkv_w, Wqkv, 1024 * 512);
    cvt_bf16_kernel<<<(512 * 512 + 255) / 256, 256, 0, stream>>>(proj_w, Wproj, 512 * 512);
    cvt_bf16_kernel<<<(2048 * 512 + 255) / 256, 256, 0, stream>>>(fc1_w, Wf1, 2048 * 512);
    cvt_bf16_kernel<<<(512 * 2048 + 255) / 256, 256, 0, stream>>>(fc2_w, Wf2, 512 * 2048);
    cvt_scale_kernel<<<(802816 + 255) / 256, 256, 0, stream>>>(qg, Q16, 802816,
                                                              0.17677669529663687f);
    bias_tab_kernel<<<(16 * 64 * 64 + 255) / 256, 256, 0, stream>>>(relTab, BT);

    // 2) LN1 + window partition -> XN bf16
    ln_win_kernel<<<25088, 256, 0, stream>>>(x, n1w, n1b, XN);

    // 3) KV projection GEMM (25088 x 1024 x 512) with head-scatter epilogue
    gemm_bf16_kernel<EPI_KV><<<dim3(196, 8), 256, 0, stream>>>(
        XN, Wqkv, 512, 1024, qkv_b, nullptr, nullptr, nullptr, K16, V16);

    // 4) windowed attention (512 windows x 16 heads)
    attn_kernel<<<8192, 128, 0, stream>>>(Q16, K16, V16, BT, AO);

    // 5) proj GEMM + window-reverse + residual -> X1 f32
    gemm_bf16_kernel<EPI_PROJ><<<dim3(196, 4), 256, 0, stream>>>(
        AO, Wproj, 512, 512, proj_b, x, X1, nullptr, nullptr, nullptr);

    // 6) LN2 -> H2 bf16 (reuses XN region)
    unsigned short* H2 = XN;
    ln_plain_kernel<<<25088, 256, 0, stream>>>(X1, n2w, n2b, H2);

    // 7) fc1 GEMM + exact GELU -> M1 bf16 (25088 x 2048 x 512)
    gemm_bf16_kernel<EPI_GELU><<<dim3(196, 16), 256, 0, stream>>>(
        H2, Wf1, 512, 2048, fc1_b, nullptr, nullptr, M1, nullptr, nullptr);

    // 8) fc2 GEMM + bias + residual -> d_out f32 (25088 x 512 x 2048)
    gemm_bf16_kernel<EPI_OUT><<<dim3(196, 4), 256, 0, stream>>>(
        M1, Wf2, 2048, 512, fc2_b, X1, out, nullptr, nullptr, nullptr);
}
